// MultiHeadAttention_27951647162545
// MI455X (gfx1250) — matrix operand order repaired and checked by reading, not compiled, not run on hardware
//
#include <hip/hip_runtime.h>
#include <hip/hip_bf16.h>

// ---------------------------------------------------------------------------
// Fused causal MHA for MI455X (gfx1250, wave32): bf16 WMMA + async global->LDS.
// B=4, T=2048, D=1024, H=16, DH=64.
//   K0a: x -> bf16                      (one-shot convert)
//   K0b: W -> bf16 transposed           (one-shot tiled transpose-convert)
//   K1 : QKV projection (async-staged, double-buffered, 12 WMMA/k-step)
//        -> Q,K bf16 [B,H,T,DH] (Q pre-scaled 1/8), V bf16 [B,H,DH,T]
//   K2 : flash attention (async-staged K/V, online softmax) -> ctx bf16 [B,T,D]
//   K3 : output projection (async-staged) -> fp32 d_out
// ---------------------------------------------------------------------------

typedef __attribute__((ext_vector_type(16))) __bf16 v16bf;
typedef __attribute__((ext_vector_type(4)))  __bf16 v4bf;
typedef __attribute__((ext_vector_type(8)))  float  v8f;

#define Bn  4
#define Tn  2048
#define Dn  1024
#define Hn  16
#define DHn 64

static __device__ __forceinline__ v8f wmma_bf16(v16bf a, v16bf b, v8f c) {
  return __builtin_amdgcn_wmma_f32_16x16x32_bf16(
      false, a, false, b, (short)0, c, false, false);
}

static __device__ __forceinline__ v8f zero8() {
  v8f z;
#pragma unroll
  for (int i = 0; i < 8; ++i) z[i] = 0.0f;
  return z;
}

// Async 16-byte global -> LDS copy (per active lane), tracked by ASYNCcnt.
static __device__ __forceinline__ void async_copy16(unsigned lds_addr,
                                                    const void* gaddr) {
  asm volatile("global_load_async_to_lds_b128 %0, %1, off"
               :: "v"(lds_addr), "v"(gaddr) : "memory");
}
static __device__ __forceinline__ void wait_async0() {
  asm volatile("s_wait_asynccnt 0x0" ::: "memory");
}
static __device__ __forceinline__ unsigned lds_off(const void* p) {
  return (unsigned)(uintptr_t)p;   // flat LDS pointer: low 32 bits = LDS addr
}

// ===========================================================================
// K0a: x fp32 -> bf16, 4 elements/thread.
// ===========================================================================
__global__ __launch_bounds__(256)
void convert_x_kernel(const float* __restrict__ x, __bf16* __restrict__ xbf) {
  const size_t i = ((size_t)blockIdx.x * 256 + threadIdx.x) * 4;
  const float4 v = *(const float4*)(x + i);
  v4bf p;
  p[0] = (__bf16)v.x; p[1] = (__bf16)v.y;
  p[2] = (__bf16)v.z; p[3] = (__bf16)v.w;
  *(v4bf*)(xbf + i) = p;
}

// ===========================================================================
// K0b: WT[n][k] = bf16(W[k][n]) for WQ/WK/WV/WO. 32x32 LDS tiles.
// grid (32, 32, 4), 256 threads.
// ===========================================================================
__global__ __launch_bounds__(256)
void transpose_w_kernel(const float* __restrict__ WQ, const float* __restrict__ WK,
                        const float* __restrict__ WV, const float* __restrict__ WO,
                        __bf16* __restrict__ WQT, __bf16* __restrict__ WKT,
                        __bf16* __restrict__ WVT, __bf16* __restrict__ WOT) {
  __shared__ float tile[32][33];
  const int tx = threadIdx.x & 31, ty = threadIdx.x >> 5;
  const int k0 = blockIdx.x * 32, n0 = blockIdx.y * 32, z = blockIdx.z;
  const float* W = (z == 0) ? WQ : (z == 1) ? WK : (z == 2) ? WV : WO;
  __bf16*     WT = (z == 0) ? WQT : (z == 1) ? WKT : (z == 2) ? WVT : WOT;
#pragma unroll
  for (int rr = 0; rr < 4; ++rr) {
    const int r = ty + (rr << 3);
    tile[r][tx] = W[(size_t)(k0 + r) * Dn + n0 + tx];
  }
  __syncthreads();
#pragma unroll
  for (int rr = 0; rr < 4; ++rr) {
    const int r = ty + (rr << 3);
    WT[(size_t)(n0 + r) * Dn + k0 + tx] = (__bf16)tile[tx][r];
  }
}

// ===========================================================================
// K1: fused QKV projection, async-staged + double-buffered.
// grid (64, 16), 256 threads (8 waves); wave w owns rows w*16..w*16+15.
// ===========================================================================
__global__ __launch_bounds__(256)
void qkv_proj_kernel(const __bf16* __restrict__ xbf,
                     const __bf16* __restrict__ WQT,
                     const __bf16* __restrict__ WKT,
                     const __bf16* __restrict__ WVT,
                     const float* __restrict__ bQ, const float* __restrict__ bK,
                     const float* __restrict__ bV,
                     __bf16* __restrict__ Qb, __bf16* __restrict__ Kb,
                     __bf16* __restrict__ Vt) {
  __shared__ __bf16 xs[2][128][40];       // 80B row stride (16B aligned)
  __shared__ __bf16 wsT[2][3][64][40];

  const int t    = threadIdx.x;
  const int w    = t >> 5;
  const int lane = t & 31;
  const int half = lane >> 4;
  const int ml   = lane & 15;
  const int m0   = blockIdx.x * 128;
  const int n0   = blockIdx.y * 64;

  auto stage = [&](int kb, int bufi) {
#pragma unroll
    for (int it = 0; it < 2; ++it) {      // x tile: 128 rows x 4 chunks
      const int i  = t + (it << 8);
      const int r  = i >> 2;
      const int c8 = (i & 3) << 3;
      async_copy16(lds_off(&xs[bufi][r][c8]),
                   xbf + (size_t)(m0 + r) * Dn + kb + c8);
    }
    for (int i = t; i < 768; i += 256) {  // 3 W tiles: 64 rows x 4 chunks
      const int mat = i >> 8;
      const int j   = i & 255;
      const int nr  = j >> 2;
      const int c8  = (j & 3) << 3;
      const __bf16* Wp = (mat == 0) ? WQT : (mat == 1) ? WKT : WVT;
      async_copy16(lds_off(&wsT[bufi][mat][nr][c8]),
                   Wp + (size_t)(n0 + nr) * Dn + kb + c8);
    }
  };

  v8f acc[3][4];
#pragma unroll
  for (int mt = 0; mt < 3; ++mt)
#pragma unroll
    for (int nt = 0; nt < 4; ++nt) acc[mt][nt] = zero8();

  stage(0, 0);
  int buf = 0;
  for (int kb = 0; kb < Dn; kb += 32) {
    wait_async0();
    __syncthreads();
    if (kb + 32 < Dn) stage(kb + 32, buf ^ 1);

    v16bf af;
    const int arow = (w << 4) + ml;
#pragma unroll
    for (int j = 0; j < 16; ++j)
      af[j] = xs[buf][arow][(half << 4) + j];
#pragma unroll
    for (int mat = 0; mat < 3; ++mat) {
#pragma unroll
      for (int nt = 0; nt < 4; ++nt) {
        v16bf bf_;
#pragma unroll
        for (int j = 0; j < 16; ++j)
          bf_[j] = wsT[buf][mat][(nt << 4) + ml][(half << 4) + j];
        acc[mat][nt] = wmma_bf16(af, bf_, acc[mat][nt]);
      }
    }
    buf ^= 1;
  }

  // epilogue: bias, Q scale, write Q/K [B,H,T,DH], V transposed [B,H,DH,T]
#pragma unroll
  for (int mat = 0; mat < 3; ++mat) {
    const float* bp = (mat == 0) ? bQ : (mat == 1) ? bK : bV;
    const float sc  = (mat == 0) ? 0.125f : 1.0f;
#pragma unroll
    for (int nt = 0; nt < 4; ++nt) {
      const int n    = n0 + (nt << 4) + ml;
      const int h    = n >> 6;
      const int dh   = n & 63;
      const float bb = bp[n];
#pragma unroll
      for (int r = 0; r < 8; ++r) {
        const int grow = m0 + (w << 4) + (half << 3) + r;
        const int bi   = grow >> 11;
        const int ti   = grow & (Tn - 1);
        const float val = (acc[mat][nt][r] + bb) * sc;
        if (mat == 2) {
          Vt[((size_t)(bi * Hn + h) * DHn + dh) * Tn + ti] = (__bf16)val;
        } else {
          __bf16* dst = (mat == 0) ? Qb : Kb;
          dst[((size_t)(bi * Hn + h) * Tn + ti) * DHn + dh] = (__bf16)val;
        }
      }
    }
  }
}

// ===========================================================================
// K2: flash attention, causal; async-staged + double-buffered K/V blocks.
// grid (T/128, H, B), 256 threads; wave w owns 16 query rows.
// ===========================================================================
__global__ __launch_bounds__(256)
void flash_attn_kernel(const __bf16* __restrict__ Qb,
                       const __bf16* __restrict__ Kb,
                       const __bf16* __restrict__ Vt,
                       __bf16* __restrict__ Ctx) {
  __shared__ __bf16 kt[2][64][72];     // [key][dh], 144B row stride
  __shared__ __bf16 vt[2][64][72];     // [dh][key]
  __shared__ __bf16 ps[8][16][72];     // per-wave P tile [row][key]

  const int t    = threadIdx.x;
  const int w    = t >> 5;
  const int lane = t & 31;
  const int half = lane >> 4;
  const int ml   = lane & 15;
  const int q0   = blockIdx.x * 128;
  const int h    = blockIdx.y;
  const int b    = blockIdx.z;
  const size_t headbase = (size_t)(b * Hn + h) * (Tn * DHn);

  auto stageKV = [&](int kb0, int bufi) {
#pragma unroll
    for (int it = 0; it < 2; ++it) {   // 64 rows x 8 chunks, for K and V
      const int i  = t + (it << 8);
      const int r  = i >> 3;
      const int c8 = (i & 7) << 3;
      async_copy16(lds_off(&kt[bufi][r][c8]),
                   Kb + headbase + (size_t)(kb0 + r) * DHn + c8);
      async_copy16(lds_off(&vt[bufi][r][c8]),
                   Vt + headbase + (size_t)r * Tn + kb0 + c8);
    }
  };

  // Q A-fragments (fixed rows for whole kernel)
  const __bf16* qrow = Qb + headbase + (size_t)(q0 + (w << 4) + ml) * DHn;
  v16bf qa0, qa1;
#pragma unroll
  for (int j = 0; j < 16; ++j) {
    const int d = (half << 4) + j;
    qa0[j] = qrow[d];
    qa1[j] = qrow[32 + d];
  }

  v8f o[4];
#pragma unroll
  for (int n = 0; n < 4; ++n) o[n] = zero8();
  float m_[8], l_[8];
#pragma unroll
  for (int r = 0; r < 8; ++r) { m_[r] = -3.0e38f; l_[r] = 0.0f; }

  const int nkb = (q0 >> 6) + 2;
  stageKV(0, 0);
  int buf = 0;
  for (int ib = 0; ib < nkb; ++ib) {
    const int kb0 = ib << 6;
    wait_async0();
    __syncthreads();
    if (ib + 1 < nkb) stageKV((ib + 1) << 6, buf ^ 1);

    if (kb0 <= q0 + (w << 4) + 15) {   // wave-uniform causal skip
      v8f s[4];
#pragma unroll
      for (int st = 0; st < 4; ++st) {
        v16bf kb_0, kb_1;
#pragma unroll
        for (int j = 0; j < 16; ++j) {
          kb_0[j] = kt[buf][(st << 4) + ml][(half << 4) + j];
          kb_1[j] = kt[buf][(st << 4) + ml][32 + (half << 4) + j];
        }
        v8f sv = zero8();
        sv = wmma_bf16(qa0, kb_0, sv);
        sv = wmma_bf16(qa1, kb_1, sv);
        s[st] = sv;
      }

      float rmax[8];
#pragma unroll
      for (int r = 0; r < 8; ++r) {
        const int qg = q0 + (w << 4) + (half << 3) + r;
        float mx = -3.0e38f;
#pragma unroll
        for (int st = 0; st < 4; ++st) {
          const int kg = kb0 + (st << 4) + ml;
          float sv = s[st][r];
          if (kg > qg) sv = -1.0e30f;
          s[st][r] = sv;
          mx = fmaxf(mx, sv);
        }
        rmax[r] = mx;
      }
#pragma unroll
      for (int r = 0; r < 8; ++r)
#pragma unroll
        for (int off = 1; off < 16; off <<= 1)
          rmax[r] = fmaxf(rmax[r], __shfl_xor(rmax[r], off, 32));

      float rsum[8];
#pragma unroll
      for (int r = 0; r < 8; ++r) {
        const float mn    = fmaxf(m_[r], rmax[r]);
        const float alpha = __expf(m_[r] - mn);
        m_[r] = mn;
        l_[r] *= alpha;
#pragma unroll
        for (int n = 0; n < 4; ++n) o[n][r] *= alpha;
        float psum = 0.0f;
#pragma unroll
        for (int st = 0; st < 4; ++st) {
          const float p = __expf(s[st][r] - mn);
          psum += p;
          ps[w][(half << 3) + r][(st << 4) + ml] = (__bf16)p;
        }
        rsum[r] = psum;
      }
#pragma unroll
      for (int r = 0; r < 8; ++r) {
#pragma unroll
        for (int off = 1; off < 16; off <<= 1)
          rsum[r] += __shfl_xor(rsum[r], off, 32);
        l_[r] += rsum[r];
      }
      __asm__ volatile("" ::: "memory");   // keep P stores before P loads

      v16bf pa0, pa1;
#pragma unroll
      for (int j = 0; j < 16; ++j) {
        const int kk = (half << 4) + j;
        pa0[j] = ps[w][ml][kk];
        pa1[j] = ps[w][ml][32 + kk];
      }
#pragma unroll
      for (int n = 0; n < 4; ++n) {
        v16bf vb0, vb1;
#pragma unroll
        for (int j = 0; j < 16; ++j) {
          vb0[j] = vt[buf][(n << 4) + ml][(half << 4) + j];
          vb1[j] = vt[buf][(n << 4) + ml][32 + (half << 4) + j];
        }
        o[n] = wmma_bf16(pa0, vb0, o[n]);
        o[n] = wmma_bf16(pa1, vb1, o[n]);
      }
    }
    buf ^= 1;
  }

  float inv[8];
#pragma unroll
  for (int r = 0; r < 8; ++r) inv[r] = 1.0f / l_[r];
#pragma unroll
  for (int n = 0; n < 4; ++n) {
    const int dcol = h * DHn + (n << 4) + ml;
#pragma unroll
    for (int r = 0; r < 8; ++r) {
      const int qg = q0 + (w << 4) + (half << 3) + r;
      Ctx[((size_t)b * Tn + qg) * Dn + dcol] = (__bf16)(o[n][r] * inv[r]);
    }
  }
}

// ===========================================================================
// K3: output projection, async-staged + double-buffered. fp32 out.
// ===========================================================================
__global__ __launch_bounds__(256)
void out_proj_kernel(const __bf16* __restrict__ Ctx,
                     const __bf16* __restrict__ WOT,
                     const float* __restrict__ bO,
                     float* __restrict__ out) {
  __shared__ __bf16 xs[2][128][40];
  __shared__ __bf16 wsT[2][64][40];

  const int t    = threadIdx.x;
  const int w    = t >> 5;
  const int lane = t & 31;
  const int half = lane >> 4;
  const int ml   = lane & 15;
  const int m0   = blockIdx.x * 128;
  const int n0   = blockIdx.y * 64;

  auto stage = [&](int kb, int bufi) {
#pragma unroll
    for (int it = 0; it < 2; ++it) {   // ctx tile: 128 rows x 4 chunks
      const int i  = t + (it << 8);
      const int r  = i >> 2;
      const int c8 = (i & 3) << 3;
      async_copy16(lds_off(&xs[bufi][r][c8]),
                   Ctx + (size_t)(m0 + r) * Dn + kb + c8);
    }
    if (t < 256) {                     // WO tile: 64 rows x 4 chunks
      const int nr = t >> 2;
      const int c8 = (t & 3) << 3;
      async_copy16(lds_off(&wsT[bufi][nr][c8]),
                   WOT + (size_t)(n0 + nr) * Dn + kb + c8);
    }
  };

  v8f acc[4];
#pragma unroll
  for (int nt = 0; nt < 4; ++nt) acc[nt] = zero8();

  stage(0, 0);
  int buf = 0;
  for (int kb = 0; kb < Dn; kb += 32) {
    wait_async0();
    __syncthreads();
    if (kb + 32 < Dn) stage(kb + 32, buf ^ 1);

    v16bf af;
    const int arow = (w << 4) + ml;
#pragma unroll
    for (int j = 0; j < 16; ++j)
      af[j] = xs[buf][arow][(half << 4) + j];
#pragma unroll
    for (int nt = 0; nt < 4; ++nt) {
      v16bf bf_;
#pragma unroll
      for (int j = 0; j < 16; ++j)
        bf_[j] = wsT[buf][(nt << 4) + ml][(half << 4) + j];
      acc[nt] = wmma_bf16(af, bf_, acc[nt]);
    }
    buf ^= 1;
  }

#pragma unroll
  for (int nt = 0; nt < 4; ++nt) {
    const int n    = n0 + (nt << 4) + ml;
    const float bb = bO[n];
#pragma unroll
    for (int r = 0; r < 8; ++r) {
      const int grow = m0 + (w << 4) + (half << 3) + r;
      out[(size_t)grow * Dn + n] = acc[nt][r] + bb;
    }
  }
}

// ===========================================================================
extern "C" void kernel_launch(void* const* d_in, const int* in_sizes, int n_in,
                              void* d_out, int out_size, void* d_ws,
                              size_t ws_size, hipStream_t stream) {
  const float* x  = (const float*)d_in[0];
  // d_in[1] = causal mask, implemented analytically
  const float* WQ = (const float*)d_in[2];
  const float* bQ = (const float*)d_in[3];
  const float* WK = (const float*)d_in[4];
  const float* bK = (const float*)d_in[5];
  const float* WV = (const float*)d_in[6];
  const float* bV = (const float*)d_in[7];
  const float* WO = (const float*)d_in[8];
  const float* bO = (const float*)d_in[9];
  float* out = (float*)d_out;

  const size_t nElem = (size_t)Bn * Tn * Dn;   // 8,388,608
  const size_t wElem = (size_t)Dn * Dn;        // 1,048,576
  char* ws = (char*)d_ws;
  size_t off = 0;
  __bf16* xbf = (__bf16*)(ws + off); off += nElem * 2;
  __bf16* WQT = (__bf16*)(ws + off); off += wElem * 2;
  __bf16* WKT = (__bf16*)(ws + off); off += wElem * 2;
  __bf16* WVT = (__bf16*)(ws + off); off += wElem * 2;
  __bf16* WOT = (__bf16*)(ws + off); off += wElem * 2;
  __bf16* Qb  = (__bf16*)(ws + off); off += nElem * 2;
  __bf16* Kb  = (__bf16*)(ws + off); off += nElem * 2;
  __bf16* Vt  = (__bf16*)(ws + off); off += nElem * 2;
  __bf16* Ctx = (__bf16*)(ws + off); off += nElem * 2;

  dim3 blk(256);
  convert_x_kernel<<<dim3(nElem / 1024), blk, 0, stream>>>(x, xbf);
  transpose_w_kernel<<<dim3(32, 32, 4), blk, 0, stream>>>(
      WQ, WK, WV, WO, WQT, WKT, WVT, WOT);
  dim3 g1((Bn * Tn) / 128, Dn / 64);
  qkv_proj_kernel<<<g1, blk, 0, stream>>>(xbf, WQT, WKT, WVT, bQ, bK, bV,
                                          Qb, Kb, Vt);
  dim3 g2(Tn / 128, Hn, Bn);
  flash_attn_kernel<<<g2, blk, 0, stream>>>(Qb, Kb, Vt, Ctx);
  out_proj_kernel<<<g1, blk, 0, stream>>>(Ctx, WOT, bO, out);
}
